// Mamba_25898652795533
// MI455X (gfx1250) — matrix operand order repaired and checked by reading, not compiled, not run on hardware
//
#include <hip/hip_runtime.h>
#include <hip/hip_bf16.h>
#include <math.h>

#define BB 2
#define LL 1024
#define DM 768
#define EDIM 1536
#define DST 16
#define DTR 48
#define NLAYERS 2
#define NVOCAB 32000
#define MROWS (BB*LL)
#define DBCW (DTR + 2*DST)   // 80

typedef _Float16 v4h  __attribute__((ext_vector_type(4)));
typedef _Float16 v8h  __attribute__((ext_vector_type(8)));
typedef _Float16 v16h __attribute__((ext_vector_type(16)));
typedef float    v8f  __attribute__((ext_vector_type(8)));

#define BM 128
#define BN 128
#define BK 32
#define LDST 40   // halves per LDS row: 32 + 8 pad -> conflict-free ds_load_b128

__device__ __forceinline__ float sigmoidf_(float x) { return 1.f / (1.f + __expf(-x)); }

// ---------------- embedding gather ----------------
__global__ __launch_bounds__(256) void embed_kernel(const int* __restrict__ ids,
                                                    const float* __restrict__ emb,
                                                    float* __restrict__ x)
{
  int idx = blockIdx.x * 256 + threadIdx.x;
  if (idx >= MROWS * DM) return;
  int d = idx % DM, bl = idx / DM;
  x[idx] = emb[(size_t)ids[bl] * DM + d];
}

// ---------------- fp32 -> f16 conversion (vectorized) ----------------
__global__ __launch_bounds__(256) void cvt_f16_kernel(const float* __restrict__ s,
                                                      _Float16* __restrict__ d, int n4)
{
  int i = blockIdx.x * 256 + threadIdx.x;
  if (i >= n4) return;
  float4 v = ((const float4*)s)[i];
  v4h h = {(_Float16)v.x, (_Float16)v.y, (_Float16)v.z, (_Float16)v.w};
  ((v4h*)d)[i] = h;
}

// ---------------- rmsnorm (one block per row), f16 output ----------------
__global__ __launch_bounds__(256) void rmsnorm_kernel(const float* __restrict__ x,
                                                      const float* __restrict__ w,
                                                      _Float16* __restrict__ o, int D)
{
  __shared__ float red[8];
  int row = blockIdx.x;
  const float* xr = x + (size_t)row * D;
  float s = 0.f;
  for (int i = threadIdx.x; i < D; i += 256) { float v = xr[i]; s += v * v; }
  #pragma unroll
  for (int off = 16; off > 0; off >>= 1) s += __shfl_xor(s, off, 32);
  if ((threadIdx.x & 31) == 0) red[threadIdx.x >> 5] = s;
  __syncthreads();
  if (threadIdx.x == 0) { float t = 0.f; for (int i = 0; i < 8; ++i) t += red[i]; red[0] = t; }
  __syncthreads();
  float scale = rsqrtf(red[0] / (float)D + 1e-5f);
  for (int i = threadIdx.x; i < D; i += 256)
    o[(size_t)row * D + i] = (_Float16)(xr[i] * scale * w[i]);
}

// ---------------- staged 16-byte (8-half) tile load ----------------
template<bool BOUNDS>
__device__ __forceinline__ uint4 ld_tile16(const _Float16* __restrict__ base,
                                           int row, int lim, int lda, int kk, int K)
{
  if (!BOUNDS) {
    return *(const uint4*)(base + (size_t)row * lda + kk);
  } else {
    bool ok = (row < lim) && ((kk + 7) < K);
    const uint4* p = (const uint4*)(base + (size_t)(ok ? row : 0) * lda + (ok ? kk : 0));
    uint4 v = *p;                  // always in-bounds (clamped), then mask
    if (!ok) { v.x = 0u; v.y = 0u; v.z = 0u; v.w = 0u; }
    return v;
  }
}

// ---------------- WMMA GEMM: C = epi(A @ W^T), f16 operands, f32 accum ----------------
// A: (M,K) f16 row-major (stride lda).  W: (N,K) f16 row-major (stride K).
enum { EPI_NONE = 0, EPI_BIAS_SOFTPLUS = 1, EPI_RESID = 2 };

template<int EPI, bool BOUNDS>
__global__ __launch_bounds__(256) void gemm_xwt_wmma(
    const _Float16* __restrict__ A, int lda,
    const _Float16* __restrict__ W,
    const float* __restrict__ aux,     // bias (EPI=1, len N) or residual (EPI=2, (M,ldc))
    float* __restrict__ C, int ldc,
    int M, int N, int K)
{
  __shared__ __align__(16) _Float16 As[2][BM * LDST];
  __shared__ __align__(16) _Float16 Ws[2][BN * LDST];

  const int tid  = threadIdx.x;
  const int lane = tid & 31;
  const int wav  = tid >> 5;
  const int lr   = lane & 15;
  const int g    = lane >> 4;
  const int wm   = wav >> 1;          // 0..3 : 32-row sub-block
  const int wn   = wav & 1;           // 0..1 : 64-col sub-block
  const int m0   = blockIdx.y * BM;
  const int n0   = blockIdx.x * BN;

  // staging geometry: 4 threads per 32-half row, 2 row-chunks of 64
  const int sr0 = tid >> 2;           // 0..63
  const int sr1 = sr0 + 64;           // 64..127
  const int sk  = (tid & 3) << 3;     // 0,8,16,24 (halves)

  const v8f vzero = {0.f, 0.f, 0.f, 0.f, 0.f, 0.f, 0.f, 0.f};
  v8f acc[2][4];
  #pragma unroll
  for (int i = 0; i < 2; ++i)
    #pragma unroll
    for (int j = 0; j < 4; ++j) acc[i][j] = vzero;

  const int nk = (K + BK - 1) / BK;

  // prologue: tile 0 -> LDS buffer 0
  uint4 ra0 = ld_tile16<BOUNDS>(A, m0 + sr0, M, lda, sk, K);
  uint4 ra1 = ld_tile16<BOUNDS>(A, m0 + sr1, M, lda, sk, K);
  uint4 rw0 = ld_tile16<BOUNDS>(W, n0 + sr0, N, K,   sk, K);
  uint4 rw1 = ld_tile16<BOUNDS>(W, n0 + sr1, N, K,   sk, K);
  *(uint4*)&As[0][sr0 * LDST + sk] = ra0;
  *(uint4*)&As[0][sr1 * LDST + sk] = ra1;
  *(uint4*)&Ws[0][sr0 * LDST + sk] = rw0;
  *(uint4*)&Ws[0][sr1 * LDST + sk] = rw1;
  __syncthreads();

  for (int kt = 0; kt < nk; ++kt) {
    const int cur = kt & 1;
    const bool more = (kt + 1) < nk;

    // issue next tile's global loads before compute (latency hiding)
    if (more) {
      int kk = (kt + 1) * BK + sk;
      ra0 = ld_tile16<BOUNDS>(A, m0 + sr0, M, lda, kk, K);
      ra1 = ld_tile16<BOUNDS>(A, m0 + sr1, M, lda, kk, K);
      rw0 = ld_tile16<BOUNDS>(W, n0 + sr0, N, K,   kk, K);
      rw1 = ld_tile16<BOUNDS>(W, n0 + sr1, N, K,   kk, K);
    }

    // ---- fragments from current buffer ----
    v16h af[2], bf[4];
    #pragma unroll
    for (int mi = 0; mi < 2; ++mi) {
      // A layout: lane (M=lr, group g); halves = K[8g..8g+7] ++ K[16+8g..16+8g+7]
      const _Float16* p = &As[cur][(wm * 32 + mi * 16 + lr) * LDST + 8 * g];
      v8h lo = *(const v8h*)p;
      v8h hi = *(const v8h*)(p + 16);
      af[mi] = __builtin_shufflevector(lo, hi, 0,1,2,3,4,5,6,7,8,9,10,11,12,13,14,15);
    }
    #pragma unroll
    for (int nj = 0; nj < 4; ++nj) {
      // B layout: lane (N=lr, group g); halves = K[8g..8g+7] ++ K[16+8g..16+8g+7]
      const _Float16* p = &Ws[cur][(wn * 64 + nj * 16 + lr) * LDST + 8 * g];
      v8h lo = *(const v8h*)p;
      v8h hi = *(const v8h*)(p + 16);
      bf[nj] = __builtin_shufflevector(lo, hi, 0,1,2,3,4,5,6,7,8,9,10,11,12,13,14,15);
    }
    #pragma unroll
    for (int mi = 0; mi < 2; ++mi)
      #pragma unroll
      for (int nj = 0; nj < 4; ++nj)
        acc[mi][nj] = __builtin_amdgcn_wmma_f32_16x16x32_f16(
            false, af[mi], false, bf[nj], (short)0, acc[mi][nj], false, false);

    // park next tile in the opposite buffer
    if (more) {
      const int nxt = cur ^ 1;
      *(uint4*)&As[nxt][sr0 * LDST + sk] = ra0;
      *(uint4*)&As[nxt][sr1 * LDST + sk] = ra1;
      *(uint4*)&Ws[nxt][sr0 * LDST + sk] = rw0;
      *(uint4*)&Ws[nxt][sr1 * LDST + sk] = rw1;
    }
    __syncthreads();
  }

  // ---- epilogue: C layout lane lr = col, vgpr r -> row (r + 8g) ----
  #pragma unroll
  for (int mi = 0; mi < 2; ++mi) {
    #pragma unroll
    for (int nj = 0; nj < 4; ++nj) {
      int col = n0 + wn * 64 + nj * 16 + lr;
      if (BOUNDS && col >= N) continue;
      #pragma unroll
      for (int r = 0; r < 8; ++r) {
        int row = m0 + wm * 32 + mi * 16 + g * 8 + r;
        if (BOUNDS && row >= M) continue;
        float v = acc[mi][nj][r];
        if (EPI == EPI_BIAS_SOFTPLUS) {
          v += aux[col];
          v = (v > 20.f) ? v : log1pf(__expf(v));
        } else if (EPI == EPI_RESID) {
          v += aux[(size_t)row * ldc + col];
        }
        C[(size_t)row * ldc + col] = v;
      }
    }
  }
}

// ---------------- causal depthwise conv (D_CONV=4) + bias + SiLU, f16 out ----------------
__global__ __launch_bounds__(256) void conv_silu_kernel(
    const float* __restrict__ xz, const float* __restrict__ cw,
    const float* __restrict__ cb, _Float16* __restrict__ xi)
{
  int idx = blockIdx.x * 256 + threadIdx.x;     // over MROWS*EDIM
  if (idx >= MROWS * EDIM) return;
  int e  = idx % EDIM;
  int bl = idx / EDIM;
  int l  = bl % LL;
  float acc = cb[e];
  #pragma unroll
  for (int k = 0; k < 4; ++k) {
    int ls = l + k - 3;
    if (ls >= 0) acc += xz[(size_t)(bl + k - 3) * (2 * EDIM) + e] * cw[e * 4 + k];
  }
  xi[idx] = (_Float16)(acc * sigmoidf_(acc));
}

// ---------------- selective scan: one lane per (b,e,n) state ----------------
__global__ __launch_bounds__(256) void scan_kernel(
    const float* __restrict__ delta, const _Float16* __restrict__ xi,
    const float* __restrict__ dbc,   const float* __restrict__ xz,
    const float* __restrict__ A_log, const float* __restrict__ skipD,
    _Float16* __restrict__ ys)
{
  int idx = blockIdx.x * 256 + threadIdx.x;     // B*ED*DST = 49152 lanes
  int n = idx & 15;
  int c = idx >> 4;                             // b*ED + e
  int e = c % EDIM;
  int b = c / EDIM;
  float Ae = -__expf(A_log[e * DST + n]);
  float Dv = skipD[e];
  float h  = 0.f;
  const size_t rbase = (size_t)b * LL;
  for (int t = 0; t < LL; ++t) {
    size_t r = rbase + t;
    float d  = delta[r * EDIM + e];
    float xv = (float)xi[r * EDIM + e];
    float Bn = dbc[r * DBCW + DTR + n];
    float Cn = dbc[r * DBCW + DTR + DST + n];
    h = __expf(d * Ae) * h + (d * xv) * Bn;
    float y = h * Cn;
    y += __shfl_xor(y, 1, 32);
    y += __shfl_xor(y, 2, 32);
    y += __shfl_xor(y, 4, 32);
    y += __shfl_xor(y, 8, 32);
    if (n == 0) {
      float zv = xz[r * (2 * EDIM) + EDIM + e];
      ys[r * EDIM + e] = (_Float16)((y + Dv * xv) * (zv * sigmoidf_(zv)));
    }
  }
}

static inline int grid4(int n) { return (n / 4 + 255) / 256; }

extern "C" void kernel_launch(void* const* d_in, const int* in_sizes, int n_in,
                              void* d_out, int out_size, void* d_ws, size_t ws_size,
                              hipStream_t stream)
{
  const int*   ids   = (const int*)d_in[0];
  const float* emb   = (const float*)d_in[1];
  const float* ln_w  = (const float*)d_in[2];
  const float* inpj  = (const float*)d_in[3];
  const float* convw = (const float*)d_in[4];
  const float* convb = (const float*)d_in[5];
  const float* xproj = (const float*)d_in[6];
  const float* dtw   = (const float*)d_in[7];
  const float* dtb   = (const float*)d_in[8];
  const float* alog  = (const float*)d_in[9];
  const float* skpD  = (const float*)d_in[10];
  const float* outpj = (const float*)d_in[11];
  const float* nfw   = (const float*)d_in[12];
  const float* lmh   = (const float*)d_in[13];
  float* out = (float*)d_out;

  // workspace carve (256B aligned chunks)
  char* base = (char*)d_ws;
  auto carve = [&](size_t bytes) -> char* {
    char* p = base;
    base += (bytes + 255) & ~(size_t)255;
    return p;
  };
  float*    x     = (float*)   carve((size_t)MROWS * DM * 4);        // residual stream
  float*    xz    = (float*)   carve((size_t)MROWS * 2 * EDIM * 4);  // in_proj output
  float*    dbc   = (float*)   carve((size_t)MROWS * DBCW * 4);      // x_proj output
  float*    delta = (float*)   carve((size_t)MROWS * EDIM * 4);      // softplus(dt)
  _Float16* hn16  = (_Float16*)carve((size_t)MROWS * DM * 2);        // normed acts (f16)
  _Float16* xi16  = (_Float16*)carve((size_t)MROWS * EDIM * 2);      // conv+silu (f16)
  _Float16* ys16  = (_Float16*)carve((size_t)MROWS * EDIM * 2);      // scan out (f16)
  _Float16* dbc16 = (_Float16*)carve((size_t)MROWS * DBCW * 2);      // dbc (f16)
  _Float16* w16   = (_Float16*)carve((size_t)NVOCAB * DM * 2);       // f16 weight scratch

  embed_kernel<<<(MROWS * DM + 255) / 256, 256, 0, stream>>>(ids, emb, x);

  for (int l = 0; l < NLAYERS; ++l) {
    rmsnorm_kernel<<<MROWS, 256, 0, stream>>>(x, ln_w + (size_t)l * DM, hn16, DM);

    cvt_f16_kernel<<<grid4(2 * EDIM * DM), 256, 0, stream>>>(
        inpj + (size_t)l * 2 * EDIM * DM, w16, 2 * EDIM * DM / 4);
    gemm_xwt_wmma<EPI_NONE, false><<<dim3((2 * EDIM) / BN, MROWS / BM), 256, 0, stream>>>(
        hn16, DM, w16, nullptr, xz, 2 * EDIM, MROWS, 2 * EDIM, DM);

    conv_silu_kernel<<<(MROWS * EDIM + 255) / 256, 256, 0, stream>>>(
        xz, convw + (size_t)l * EDIM * 4, convb + (size_t)l * EDIM, xi16);

    cvt_f16_kernel<<<grid4(DBCW * EDIM), 256, 0, stream>>>(
        xproj + (size_t)l * DBCW * EDIM, w16, DBCW * EDIM / 4);
    gemm_xwt_wmma<EPI_NONE, true><<<dim3(1, MROWS / BM), 256, 0, stream>>>(
        xi16, EDIM, w16, nullptr, dbc, DBCW, MROWS, DBCW, EDIM);

    cvt_f16_kernel<<<grid4(MROWS * DBCW), 256, 0, stream>>>(dbc, dbc16, MROWS * DBCW / 4);
    cvt_f16_kernel<<<grid4(EDIM * DTR), 256, 0, stream>>>(
        dtw + (size_t)l * EDIM * DTR, w16, EDIM * DTR / 4);
    gemm_xwt_wmma<EPI_BIAS_SOFTPLUS, true><<<dim3(EDIM / BN, MROWS / BM), 256, 0, stream>>>(
        dbc16, DBCW, w16, dtb + (size_t)l * EDIM, delta, EDIM, MROWS, EDIM, DTR);

    scan_kernel<<<(BB * EDIM * DST) / 256, 256, 0, stream>>>(
        delta, xi16, dbc, xz, alog + (size_t)l * EDIM * DST, skpD + (size_t)l * EDIM, ys16);

    cvt_f16_kernel<<<grid4(DM * EDIM), 256, 0, stream>>>(
        outpj + (size_t)l * DM * EDIM, w16, DM * EDIM / 4);
    gemm_xwt_wmma<EPI_RESID, false><<<dim3(DM / BN, MROWS / BM), 256, 0, stream>>>(
        ys16, EDIM, w16, x, x, DM, MROWS, DM, EDIM);
  }

  rmsnorm_kernel<<<MROWS, 256, 0, stream>>>(x, nfw, hn16, DM);

  cvt_f16_kernel<<<grid4(NVOCAB * DM), 256, 0, stream>>>(lmh, w16, NVOCAB * DM / 4);
  gemm_xwt_wmma<EPI_NONE, false><<<dim3(NVOCAB / BN, MROWS / BM), 256, 0, stream>>>(
      hn16, DM, w16, nullptr, out, NVOCAB, MROWS, NVOCAB, DM);
}
